// _EfficientAttention_89026082111868
// MI455X (gfx1250) — matrix operand order repaired and checked by reading, not compiled, not run on hardware
//
#include <hip/hip_runtime.h>

// Efficient (linear) attention for MI455X / gfx1250, wave32 + WMMA bf16.
// N=16, C=768, L=8192, H=12, D=C/H=64.
//
// Memory-bound problem (~1.2 GB min traffic @ 23.3 TB/s ~= 52us floor).
// Matmuls done with V_WMMA_F32_16X16X32_BF16 using hi/lo bf16 split
// (3 WMMAs per product, f32 accumulate) for near-f32 precision while
// staying far under the compute roofline.

#define NBATCH 16
#define HEADS  12
#define DHEAD  64
#define LSEQ   8192
#define NH     (NBATCH * HEADS)               // 192 (n,h) pairs
#define CTX_STRIDE (DHEAD * DHEAD + DHEAD)    // 64x64 ctx + 64 rowsums per nh
#define LSTR   72                             // LDS row stride in bf16 elems (16B aligned, bank-spread)

typedef __attribute__((ext_vector_type(16))) __bf16 v16bf;
typedef __attribute__((ext_vector_type(8)))  float  v8f;

// ---------- bf16 split helpers (RNE) ----------
__device__ __forceinline__ unsigned short f2bf(float x) {
  union { float f; unsigned u; } v; v.f = x;
  unsigned r = v.u + 0x7FFFu + ((v.u >> 16) & 1u);
  return (unsigned short)(r >> 16);
}
__device__ __forceinline__ float bf2f(unsigned short b) {
  union { unsigned u; float f; } v; v.u = ((unsigned)b) << 16; return v.f;
}
__device__ __forceinline__ void split_bf(float x, unsigned short& hi, unsigned short& lo) {
  hi = f2bf(x);
  lo = f2bf(x - bf2f(hi));
}
// pack two adjacent elements (idx must be even) into hi/lo LDS arrays
__device__ __forceinline__ void stage_pair(float a0, float a1,
                                           unsigned short* hiArr, unsigned short* loArr, int idx) {
  unsigned short h0, l0, h1, l1;
  split_bf(a0, h0, l0); split_bf(a1, h1, l1);
  *(unsigned*)&hiArr[idx] = (unsigned)h0 | ((unsigned)h1 << 16);
  *(unsigned*)&loArr[idx] = (unsigned)l0 | ((unsigned)l1 << 16);
}

// ---------- WMMA fragment loads from row-major LDS (row = M/N index, contiguous K) ----------
union FragU { v16bf v; uint4 q[2]; };

// A 16x32 bf16: lanes 0-15 hold row M, K {0..7}+{16..23}; lanes 16-31 K {8..15}+{24..31}
__device__ __forceinline__ v16bf load_frag_A(const unsigned short* lds, int row, int kbase, int lane) {
  FragU f;
  const unsigned short* p = lds + row * LSTR + kbase + ((lane >> 4) << 3);
  f.q[0] = *(const uint4*)(p);
  f.q[1] = *(const uint4*)(p + 16);
  return f.v;
}
// B 32x16 bf16: lanes 0-15 hold col N, K {0..7}+{16..23}; lanes 16-31 K {8..15}+{24..31}
__device__ __forceinline__ v16bf load_frag_B(const unsigned short* lds, int row, int kbase, int lane) {
  FragU f;
  const unsigned short* p = lds + row * LSTR + kbase + ((lane >> 4) << 3);
  f.q[0] = *(const uint4*)(p);
  f.q[1] = *(const uint4*)(p + 16);
  return f.v;
}

__device__ __forceinline__ v8f wmma_bf16(v16bf a, v16bf b, v8f c) {
  // (neg_a, A, neg_b, B, c_mod, C, reuse_a, reuse_b)
  return __builtin_amdgcn_wmma_f32_16x16x32_bf16(false, a, false, b, (short)0, c, false, false);
}

// ---------- kernel 0: zero workspace (re-run every call; graph-replay safe) ----------
__global__ void eff_attn_zero(float* __restrict__ ws, int n) {
  int i = blockIdx.x * 256 + threadIdx.x;
  if (i < n) ws[i] = 0.0f;
}

// ---------- kernel 1: unnormalized context + key-softmax row sums ----------
// context'[d][e] = sum_l exp(k[d][l]) * k_raw[e][l] ; rowsum[d] = sum_l exp(k[d][l])
#define K1_SPLIT 8
#define K1_LC    (LSEQ / K1_SPLIT)   // 1024 L per block
#define K1_KT    64                  // inner K tile

__global__ __launch_bounds__(256) void eff_attn_context(
    const float* __restrict__ keyf, float* __restrict__ ws)
{
  __shared__ __align__(16) unsigned short sExpHi[DHEAD * LSTR];
  __shared__ __align__(16) unsigned short sExpLo[DHEAD * LSTR];
  __shared__ __align__(16) unsigned short sValHi[DHEAD * LSTR];
  __shared__ __align__(16) unsigned short sValLo[DHEAD * LSTR];
  __shared__ float sRow[DHEAD];

  const int t    = threadIdx.x;
  const int lane = t & 31;
  const int wave = t >> 5;
  const int nh   = blockIdx.x;
  const size_t l0 = (size_t)blockIdx.y * K1_LC;

  const int r  = t >> 2;          // channel row this thread loads (0..63)
  const int cb = (t & 3) * 16;    // 16 contiguous cols within the 64-wide tile

  if (t < DHEAD) sRow[t] = 0.0f;

  // wave -> output tiles: dt = wave>>1, e-tiles {2*(wave&1), 2*(wave&1)+1}
  const int dt   = wave >> 1;
  const int e0   = (wave & 1) * 2;
  const int mrow = lane & 15;

  v8f acc0 = {0.f,0.f,0.f,0.f,0.f,0.f,0.f,0.f};
  v8f acc1 = {0.f,0.f,0.f,0.f,0.f,0.f,0.f,0.f};
  float rsum = 0.0f;

  const float* src = keyf + (size_t)(nh * DHEAD + r) * LSEQ + l0 + cb;

  for (int tile = 0; tile < K1_LC / K1_KT; ++tile) {
    if (tile + 1 < K1_LC / K1_KT)
      __builtin_prefetch(src + (tile + 1) * K1_KT, 0, 3);   // global_prefetch_b8
#pragma unroll
    for (int i = 0; i < 4; ++i) {
      const float4 v4 = *(const float4*)(src + tile * K1_KT + i * 4);
      const int idx = r * LSTR + cb + i * 4;
      float ea = __expf(v4.x), eb = __expf(v4.y);
      rsum += ea + eb;
      stage_pair(ea, eb, sExpHi, sExpLo, idx);
      stage_pair(v4.x, v4.y, sValHi, sValLo, idx);
      ea = __expf(v4.z); eb = __expf(v4.w);
      rsum += ea + eb;
      stage_pair(ea, eb, sExpHi, sExpLo, idx + 2);
      stage_pair(v4.z, v4.w, sValHi, sValLo, idx + 2);
    }
    __syncthreads();
#pragma unroll
    for (int ks = 0; ks < 2; ++ks) {
      const int kb = ks * 32;
      const v16bf ahi = load_frag_A(sExpHi, dt * 16 + mrow, kb, lane);
      const v16bf alo = load_frag_A(sExpLo, dt * 16 + mrow, kb, lane);
      const v16bf b0h = load_frag_B(sValHi, (e0 + 0) * 16 + mrow, kb, lane);
      const v16bf b0l = load_frag_B(sValLo, (e0 + 0) * 16 + mrow, kb, lane);
      const v16bf b1h = load_frag_B(sValHi, (e0 + 1) * 16 + mrow, kb, lane);
      const v16bf b1l = load_frag_B(sValLo, (e0 + 1) * 16 + mrow, kb, lane);
      acc0 = wmma_bf16(ahi, b0h, acc0);
      acc0 = wmma_bf16(ahi, b0l, acc0);
      acc0 = wmma_bf16(alo, b0h, acc0);
      acc1 = wmma_bf16(ahi, b1h, acc1);
      acc1 = wmma_bf16(ahi, b1l, acc1);
      acc1 = wmma_bf16(alo, b1h, acc1);
    }
    __syncthreads();
  }

  atomicAdd(&sRow[r], rsum);
  __syncthreads();

  float* wsBase = ws + (size_t)nh * CTX_STRIDE;
  if (t < DHEAD) atomicAdd(&wsBase[DHEAD * DHEAD + t], sRow[t]);

  // C/D layout: VGPR i -> M = i + 8*(lane>=16), N = lane&15
  const int drow  = dt * 16 + ((lane >> 4) << 3);
  const int ecolA = (e0 + 0) * 16 + (lane & 15);
  const int ecolB = (e0 + 1) * 16 + (lane & 15);
#pragma unroll
  for (int i = 0; i < 8; ++i) {
    atomicAdd(&wsBase[(drow + i) * DHEAD + ecolA], acc0[i]);
    atomicAdd(&wsBase[(drow + i) * DHEAD + ecolB], acc1[i]);
  }
}

// ---------- kernel 2: query softmax + attended = ctx_norm^T * qsoft + residual ----------
#define K2_LT 128   // L tile per block

__global__ __launch_bounds__(256) void eff_attn_attend(
    const float* __restrict__ qf, const float* __restrict__ ws, float* __restrict__ out)
{
  __shared__ __align__(16) unsigned short sCtxHi[DHEAD * LSTR];   // ctx^T: rows=e, K=d
  __shared__ __align__(16) unsigned short sCtxLo[DHEAD * LSTR];
  __shared__ __align__(16) unsigned short sQHi[K2_LT * LSTR];     // q^T: rows=l, K=d
  __shared__ __align__(16) unsigned short sQLo[K2_LT * LSTR];
  __shared__ float sSum[2][K2_LT];

  const int t    = threadIdx.x;
  const int lane = t & 31;
  const int wave = t >> 5;
  const int nh   = blockIdx.x;
  const size_t l0 = (size_t)blockIdx.y * K2_LT;

  const float* wsBase = ws + (size_t)nh * CTX_STRIDE;

  // stage normalized context, transposed: sCtx[e][d] = ctx[d][e] / rowsum[d]
  {
    const int d  = t >> 2;
    const int eb = (t & 3) * 16;
    const float inv = 1.0f / wsBase[DHEAD * DHEAD + d];
#pragma unroll
    for (int i = 0; i < 16; ++i) {
      unsigned short h, l;
      split_bf(wsBase[d * DHEAD + eb + i] * inv, h, l);
      sCtxHi[(eb + i) * LSTR + d] = h;
      sCtxLo[(eb + i) * LSTR + d] = l;
    }
  }

  // query tile load + column softmax over d (64 values per l), stage transposed
  const int c    = t & (K2_LT - 1);   // column l within tile
  const int half = t >> 7;            // which 32 of the 64 rows
  const float* qb = qf + (size_t)(nh * DHEAD + half * 32) * LSEQ + l0 + c;
  float ev[32];
  float ps = 0.0f;
#pragma unroll
  for (int rr = 0; rr < 32; ++rr) {
    const float e = __expf(qb[(size_t)rr * LSEQ]);
    ev[rr] = e; ps += e;
  }
  sSum[half][c] = ps;
  __syncthreads();
  const float inv = 1.0f / (sSum[0][c] + sSum[1][c]);
#pragma unroll
  for (int rr = 0; rr < 32; rr += 2)
    stage_pair(ev[rr] * inv, ev[rr + 1] * inv, sQHi, sQLo, c * LSTR + half * 32 + rr);
  __syncthreads();

  // wave -> et = wave>>1, l-tiles (wave&1)*4 + {0..3}
  const int et   = wave >> 1;
  const int mrow = lane & 15;
  const int rowA = et * 16 + mrow;
  const v16bf a0h = load_frag_A(sCtxHi, rowA, 0,  lane);
  const v16bf a0l = load_frag_A(sCtxLo, rowA, 0,  lane);
  const v16bf a1h = load_frag_A(sCtxHi, rowA, 32, lane);
  const v16bf a1l = load_frag_A(sCtxLo, rowA, 32, lane);

#pragma unroll
  for (int j = 0; j < 4; ++j) {
    const int lt   = (wave & 1) * 4 + j;
    const int rowB = lt * 16 + mrow;
    v8f acc = {0.f,0.f,0.f,0.f,0.f,0.f,0.f,0.f};
    v16bf bh = load_frag_B(sQHi, rowB, 0, lane);
    v16bf bl = load_frag_B(sQLo, rowB, 0, lane);
    acc = wmma_bf16(a0h, bh, acc);
    acc = wmma_bf16(a0h, bl, acc);
    acc = wmma_bf16(a0l, bh, acc);
    bh = load_frag_B(sQHi, rowB, 32, lane);
    bl = load_frag_B(sQLo, rowB, 32, lane);
    acc = wmma_bf16(a1h, bh, acc);
    acc = wmma_bf16(a1h, bl, acc);
    acc = wmma_bf16(a1l, bh, acc);

    // epilogue: out = attended + query residual
    const int eM = et * 16 + ((lane >> 4) << 3);
    const int lN = lt * 16 + (lane & 15);
#pragma unroll
    for (int i = 0; i < 8; ++i) {
      const size_t off = (size_t)(nh * DHEAD + eM + i) * LSEQ + l0 + lN;
      out[off] = acc[i] + qf[off];
    }
  }
}

extern "C" void kernel_launch(void* const* d_in, const int* in_sizes, int n_in,
                              void* d_out, int out_size, void* d_ws, size_t ws_size,
                              hipStream_t stream)
{
  (void)in_sizes; (void)n_in; (void)out_size; (void)ws_size;
  const float* qf = (const float*)d_in[0];   // query_feats
  const float* kf = (const float*)d_in[1];   // key_feats
  float* out = (float*)d_out;
  float* ws  = (float*)d_ws;                 // NH * (64*64 + 64) f32 = ~3.2 MB

  const int wsElems = NH * CTX_STRIDE;
  eff_attn_zero<<<(wsElems + 255) / 256, 256, 0, stream>>>(ws, wsElems);
  eff_attn_context<<<dim3(NH, K1_SPLIT), 256, 0, stream>>>(kf, ws);
  eff_attn_attend<<<dim3(NH, LSEQ / K2_LT), 256, 0, stream>>>(qf, ws, out);
}
